// AxialAttention3DSpaceTime_73650099191895
// MI455X (gfx1250) — matrix operand order repaired and checked by reading, not compiled, not run on hardware
//
#include <hip/hip_runtime.h>

// Problem constants (from reference setup_inputs)
#define BB     2
#define TT     8
#define DD     16
#define HH     16
#define WW     16
#define F      256
#define HEADS  8
#define HD     32
#define SMAX   16
#define GSLAB  4       // 16-row slabs per block (64 tokens)
#define NROWS  64      // GSLAB * 16
#define LDSTR  260     // fp32 LDS row stride (4-dword pad; TDM pad-compatible)
#define AST    264     // f16 LDS row stride in halves (528B rows, 16B aligned)
#define PSTR   17      // padded stride for 16x16 score tiles

typedef float    v2f  __attribute__((ext_vector_type(2)));
typedef float    v8f  __attribute__((ext_vector_type(8)));
typedef _Float16 v8h  __attribute__((ext_vector_type(8)));
typedef _Float16 v16h __attribute__((ext_vector_type(16)));
typedef unsigned int u32x4 __attribute__((ext_vector_type(4)));
typedef int          i32x4 __attribute__((ext_vector_type(4)));
typedef int          i32x8 __attribute__((ext_vector_type(8)));

// LDS layout (dynamic): [Q|K|V fp32 64x260] [Xh|Xl f16 64x264] [Ps 32x(16x17) f32]
#define QKV_BYTES   (3 * NROWS * LDSTR * 4)          // 199680
#define XHL_BYTES   (2 * NROWS * AST * 2)            // 67584
#define PS_BYTES    (32 * SMAX * PSTR * 4)           // 34816
#define LDS_BYTES   (QKV_BYTES + XHL_BYTES + PS_BYTES)

static __device__ __forceinline__ v8f wmma_k4(v2f a, v2f b, v8f c) {
  return __builtin_amdgcn_wmma_f32_16x16x4_f32(false, a, false, b, (short)0, c, false, false);
}
static __device__ __forceinline__ v8f wmma_k32(v16h a, v16h b, v8f c) {
  return __builtin_amdgcn_wmma_f32_16x16x32_f16(false, a, false, b, (short)0, c, false, false);
}
static __device__ __forceinline__ v16h cat8(v8h a, v8h b) {
  return __builtin_shufflevector(a, b, 0,1,2,3,4,5,6,7,8,9,10,11,12,13,14,15);
}

// ---- Tensor Data Mover: 3D tile (256 f32 x rows x nz) global -> LDS with
// 4-DWORD LDS padding per 256 DWORDs (reproduces LDSTR=260 rows).
static __device__ __forceinline__ void tdm_load_x3d(const float* gsrc, void* lds_dst,
                                                    int rows, int nz,
                                                    long long rowStride, long long zStride)
{
  const unsigned long long ga = (unsigned long long)(uintptr_t)gsrc;
  const unsigned int       la = (unsigned int)(uintptr_t)lds_dst;
  u32x4 g0;
  g0[0] = 1u;                                      // count=1 (valid user descriptor)
  g0[1] = la;                                      // lds_addr
  g0[2] = (unsigned int)ga;                        // global_addr[31:0]
  g0[3] = (unsigned int)(ga >> 32) | (2u << 30);   // global_addr[56:32] | type=2
  i32x8 g1;
  // data_size=4B | pad_enable | pad_interval=256dw | pad_amount=4dw
  g1[0] = (int)((2u << 16) | (1u << 20) | (7u << 22) | (3u << 25));
  g1[1] = (int)(((unsigned)F & 0xFFFFu) << 16);        // tensor_dim0 lo16 @ 63:48
  g1[2] = (int)(((unsigned)rows & 0xFFFFu) << 16);     // tensor_dim1 lo16 @ 111:80
  g1[3] = (int)(((unsigned)F) << 16);                  // tile_dim0 @ 127:112
  g1[4] = (int)((unsigned)rows | ((unsigned)nz << 16)); // tile_dim1 | tile_dim2
  g1[5] = (int)(unsigned int)rowStride;                // tensor_dim0_stride[31:0]
  g1[6] = (int)(((unsigned int)(rowStride >> 32) & 0xFFFFu) |
                (((unsigned int)zStride & 0xFFFFu) << 16)); // d0s[47:32] | d1s[15:0]
  g1[7] = (int)(unsigned int)(zStride >> 16);          // tensor_dim1_stride[47:16]
  i32x4 g2;
  g2[0] = nz;                                          // tensor_dim2 (exact OOB extent)
  g2[1] = 0; g2[2] = 0; g2[3] = 0;                     // no dim3 / no iterate
  i32x4 g3 = {0, 0, 0, 0};
#if defined(__has_include)
#  if __has_include(<hip/amd_detail/amd_gfx1250_TDM.h>)
  i32x8 gz8 = {0, 0, 0, 0, 0, 0, 0, 0};
  __builtin_amdgcn_tensor_load_to_lds(g0, g1, g2, g3, gz8, 0);
#  else
  __builtin_amdgcn_tensor_load_to_lds(g0, g1, g2, g3, 0);
#  endif
#else
  __builtin_amdgcn_tensor_load_to_lds(g0, g1, g2, g3, 0);
#endif
}

// ---- Weight pre-pack: f32 [which][256][256] -> f16 hi/lo fragments in WMMA B-order.
__global__ __launch_bounds__(256, 4)
void pack_weights(const float* __restrict__ W, _Float16* __restrict__ dst, int ntiles)
{
  const int tid = blockIdx.x * blockDim.x + threadIdx.x;
  if (tid >= ntiles * 8 * 512) return;
  const int i   = tid & 15;
  const int L   = (tid >> 4) & 31;
  const int kc8 = (tid >> 9) & 7;
  const int t   = tid >> 12;
  const int which = t >> 4;
  const int k = kc8 * 32 + (L >> 4) * 16 + i;
  const int g = (t & 15) * 16 + (L & 15);
  const float v = W[((long long)which * F + k) * F + g];
  const _Float16 hi = (_Float16)v;
  const _Float16 lo = (_Float16)(v - (float)hi);
  const long long o = (long long)(t * 8 + kc8) * 1024 + L * 16 + i;
  dst[o]       = hi;
  dst[o + 512] = lo;
}

// Row r -> byte-element offset within a slab.  PACK2 packs two S=8 t-sequences
// per 16-row slab (rows 0-7 seq A, rows 8-15 seq B at +sW).
template <int PACK2>
static __device__ __forceinline__ long long rowoff(int r, long long ss) {
  return PACK2 ? ((long long)(r >> 3) * F + (long long)(r & 7) * ss)
               : (long long)r * ss;
}

// One block = GSLAB slabs of 16 tokens. 256 threads = 8 wave32.
// MODE 0: out0 = in + mha(in), out1 = same. MODE 1: out0 += mha(in).
template <int PACK2, int MODE>
__global__ __launch_bounds__(256, 1)
void axial_mha(const float* __restrict__ xin,
               const _Float16* __restrict__ wqkv_pk, const float* __restrict__ bqkv,
               const _Float16* __restrict__ wo_pk,   const float* __restrict__ bo,
               float* __restrict__ out0, float* __restrict__ out1,
               long long ss,   // token stride within a sequence
               long long is,   // stride between slabs
               long long zs,   // TDM z stride
               int e1, int e2, int e3,
               long long s0, long long s1, long long s2, long long s3)
{
  const int tid  = threadIdx.x;
  const int lane = tid & 31;
  const int wave = tid >> 5;
  const int m16  = lane & 15;
  const int kh   = lane >> 4;

  int bid = blockIdx.x;
  const int i3 = bid % e3; bid /= e3;
  const int i2 = bid % e2; bid /= e2;
  const int i1 = bid % e1; bid /= e1;
  const int i0 = bid;
  const long long base = (long long)i0 * s0 + (long long)i1 * s1 +
                         (long long)i2 * s2 + (long long)i3 * s3;

  extern __shared__ char smem[];
  float*    Qs = (float*)smem;                  // also TDM staging target for X
  float*    Ksh = Qs + NROWS * LDSTR;
  float*    Vsh = Ksh + NROWS * LDSTR;
  _Float16* Xh = (_Float16*)(smem + QKV_BYTES); // X hi, later attention-out O hi
  _Float16* Xl = Xh + NROWS * AST;
  float*    Ps = (float*)(smem + QKV_BYTES + XHL_BYTES);

  // ---- Stage X (64 rows x 256 f32) via TDM into the Q region ----
  if (wave == 0) {
    tdm_load_x3d(xin + base, (void*)Qs,
                 PACK2 ? 8 : 16, PACK2 ? 8 : 4, ss, zs);
    __builtin_amdgcn_s_wait_tensorcnt(0);
  }
  __syncthreads();

  // ---- Convert X -> f16 hi/lo (Q region is consumed here, re-used by QKV GEMM) ----
  for (int idx = tid; idx < NROWS * F; idx += 256) {
    const int r = idx >> 8;
    const int f = idx & (F - 1);
    const float v = Qs[r * LDSTR + f];
    const _Float16 h = (_Float16)v;
    Xh[r * AST + f] = h;
    Xl[r * AST + f] = (_Float16)(v - (float)h);
  }
  __syncthreads();

  // ---- QKV projection: [64 x 768]; B fragments loaded once, reused over 4 slabs ----
  for (int ct = wave; ct < 48; ct += 8) {
    const int which = ct >> 4;
    const int gc    = (ct & 15) << 4;
    const _Float16* bp = wqkv_pk + (long long)(ct * 8) * 1024 + lane * 16;
    __builtin_prefetch(bp, 0, 3);

    v8f acc[GSLAB] = {};
#pragma unroll
    for (int kc8 = 0; kc8 < 8; ++kc8) {
      const v16h Bh = *(const v16h*)(bp);
      const v16h Bl = *(const v16h*)(bp + 512);
      bp += 1024;
#pragma unroll
      for (int g = 0; g < GSLAB; ++g) {
        const _Float16* ph = Xh + (g * 16 + m16) * AST + kc8 * 32 + kh * 16;
        const _Float16* pl = Xl + (g * 16 + m16) * AST + kc8 * 32 + kh * 16;
        const v16h Ah = cat8(*(const v8h*)ph, *(const v8h*)(ph + 8));
        const v16h Al = cat8(*(const v8h*)pl, *(const v8h*)(pl + 8));
        acc[g] = wmma_k32(Ah, Bh, acc[g]);
        acc[g] = wmma_k32(Al, Bh, acc[g]);
        acc[g] = wmma_k32(Ah, Bl, acc[g]);
      }
    }
    const float bias = bqkv[which * F + gc + m16];
    float* DST = (which == 0) ? Qs : (which == 1) ? Ksh : Vsh;
#pragma unroll
    for (int g = 0; g < GSLAB; ++g)
#pragma unroll
      for (int j = 0; j < 8; ++j)
        DST[(g * 16 + kh * 8 + j) * LDSTR + gc + m16] = acc[g][j] + bias;
  }
  __syncthreads();

  // ---- Scores (exact fp32 wmma): wave = head, over all 4 slabs ----
  {
    const int hc = wave * HD;
#pragma unroll
    for (int g = 0; g < GSLAB; ++g) {
      v8f c = {};
#pragma unroll
      for (int k0 = 0; k0 < HD; k0 += 4) {
        v2f a, b;
        a.x = Qs[(g * 16 + m16) * LDSTR + hc + k0 + kh * 2 + 0];
        a.y = Qs[(g * 16 + m16) * LDSTR + hc + k0 + kh * 2 + 1];
        b.x = Ksh[(g * 16 + m16) * LDSTR + hc + k0 + kh * 2 + 0];
        b.y = Ksh[(g * 16 + m16) * LDSTR + hc + k0 + kh * 2 + 1];
        c = wmma_k4(a, b, c);
      }
      const float scale = 0.17677669529663687f;  // 1/sqrt(32)
      float* P = Ps + (g * 8 + wave) * (SMAX * PSTR);
#pragma unroll
      for (int j = 0; j < 8; ++j)
        P[(kh * 8 + j) * PSTR + m16] = c[j] * scale;
    }
  }
  __syncthreads();

  // ---- Softmax: 512 rows over 256 threads; PACK2 masks to the row's 8-key half ----
  for (int rr = tid; rr < 32 * SMAX; rr += 256) {
    float* P = Ps + (rr >> 4) * (SMAX * PSTR) + (rr & 15) * PSTR;
    const int t0 = PACK2 ? (rr & 8) : 0;
    const int t1 = PACK2 ? (t0 + 8) : SMAX;
    float mx = -3.0e38f;
    for (int t = t0; t < t1; ++t) mx = fmaxf(mx, P[t]);
    float sum = 0.0f;
    for (int t = t0; t < t1; ++t) { const float e = __expf(P[t] - mx); P[t] = e; sum += e; }
    const float inv = 1.0f / sum;
    for (int t = 0; t < SMAX; ++t)
      P[t] = (t >= t0 && t < t1) ? P[t] * inv : 0.0f;
  }
  __syncthreads();

  // ---- O = P * V (fp32 wmma); store O hi/lo over Xh/Xl ----
  {
    const int hc = wave * HD;
#pragma unroll
    for (int g = 0; g < GSLAB; ++g) {
      const float* P = Ps + (g * 8 + wave) * (SMAX * PSTR);
#pragma unroll
      for (int dt = 0; dt < 2; ++dt) {
        v8f c = {};
#pragma unroll
        for (int k0 = 0; k0 < SMAX; k0 += 4) {
          v2f a, b;
          a.x = P[m16 * PSTR + k0 + kh * 2 + 0];
          a.y = P[m16 * PSTR + k0 + kh * 2 + 1];
          b.x = Vsh[(g * 16 + k0 + kh * 2 + 0) * LDSTR + hc + dt * 16 + m16];
          b.y = Vsh[(g * 16 + k0 + kh * 2 + 1) * LDSTR + hc + dt * 16 + m16];
          c = wmma_k4(a, b, c);
        }
#pragma unroll
        for (int j = 0; j < 8; ++j) {
          const int row = g * 16 + kh * 8 + j;
          const int col = hc + dt * 16 + m16;
          const float v = c[j];
          const _Float16 h2 = (_Float16)v;
          Xh[row * AST + col] = h2;
          Xl[row * AST + col] = (_Float16)(v - (float)h2);
        }
      }
    }
  }
  __syncthreads();

  // ---- Output projection + fused bias / residual epilogue ----
  for (int ct = wave; ct < 16; ct += 8) {
    const int gc = ct << 4;
    const _Float16* bp = wo_pk + (long long)(ct * 8) * 1024 + lane * 16;
    __builtin_prefetch(bp, 0, 3);

    v8f acc[GSLAB] = {};
#pragma unroll
    for (int kc8 = 0; kc8 < 8; ++kc8) {
      const v16h Bh = *(const v16h*)(bp);
      const v16h Bl = *(const v16h*)(bp + 512);
      bp += 1024;
#pragma unroll
      for (int g = 0; g < GSLAB; ++g) {
        const _Float16* ph = Xh + (g * 16 + m16) * AST + kc8 * 32 + kh * 16;
        const _Float16* pl = Xl + (g * 16 + m16) * AST + kc8 * 32 + kh * 16;
        const v16h Ah = cat8(*(const v8h*)ph, *(const v8h*)(ph + 8));
        const v16h Al = cat8(*(const v8h*)pl, *(const v8h*)(pl + 8));
        acc[g] = wmma_k32(Ah, Bh, acc[g]);
        acc[g] = wmma_k32(Al, Bh, acc[g]);
        acc[g] = wmma_k32(Ah, Bl, acc[g]);
      }
    }
    const float bias = bo[gc + m16];
#pragma unroll
    for (int g = 0; g < GSLAB; ++g)
#pragma unroll
      for (int j = 0; j < 8; ++j) {
        const int r = kh * 8 + j;
        const long long idx = base + (long long)g * is + rowoff<PACK2>(r, ss) + gc + m16;
        const float y = acc[g][j] + bias;
        if (MODE == 0) {
          const float v = xin[idx] + y;
          out0[idx] = v;
          out1[idx] = v;
        } else {
          out0[idx] += y;
        }
      }
  }
}

extern "C" void kernel_launch(void* const* d_in, const int* in_sizes, int n_in,
                              void* d_out, int out_size, void* d_ws, size_t ws_size,
                              hipStream_t stream)
{
  const float* x = (const float*)d_in[0];
  const float* wqkv[4] = {(const float*)d_in[1], (const float*)d_in[5],
                          (const float*)d_in[9], (const float*)d_in[13]};
  const float* bqkv[4] = {(const float*)d_in[2], (const float*)d_in[6],
                          (const float*)d_in[10], (const float*)d_in[14]};
  const float* wo[4]   = {(const float*)d_in[3], (const float*)d_in[7],
                          (const float*)d_in[11], (const float*)d_in[15]};
  const float* bo[4]   = {(const float*)d_in[4], (const float*)d_in[8],
                          (const float*)d_in[12], (const float*)d_in[16]};

  float* out = (float*)d_out;
  char*  wsb = (char*)d_ws;
  float* x1  = (float*)wsb;                                  // 64 MiB
  _Float16* pkbase = (_Float16*)(wsb + (size_t)64 * 1024 * 1024);
  const long long QKV_PK = 48LL * 8 * 1024;
  const long long WO_PK  = 16LL * 8 * 1024;
  _Float16* qkv_pk[4];
  _Float16* wo_pk[4];
  for (int a = 0; a < 4; ++a) {
    qkv_pk[a] = pkbase + a * (QKV_PK + WO_PK);
    wo_pk[a]  = qkv_pk[a] + QKV_PK;
  }

  for (int a = 0; a < 4; ++a) {
    pack_weights<<<(48 * 8 * 512) / 256, 256, 0, stream>>>(wqkv[a], qkv_pk[a], 48);
    pack_weights<<<(16 * 8 * 512) / 256, 256, 0, stream>>>(wo[a], wo_pk[a], 16);
  }

  // Element strides of logical [b, t, d, h, w, f]
  const long long sW = F;
  const long long sH = (long long)WW * F;
  const long long sD = (long long)HH * WW * F;
  const long long sT = (long long)DD * HH * WW * F;
  const long long sB = (long long)TT * DD * HH * WW * F;
  const size_t lds = (size_t)LDS_BYTES;

  // t-pass (PACK2): 8 t-sequences (4 packed slabs) along w per block.
  // outer (b, d, h, w/8); TDM z = 8 steps of sW, rows = 8 of sT.
  axial_mha<1, 0><<<BB * DD * HH * (WW / 8), 256, lds, stream>>>(
      x, qkv_pk[0], bqkv[0], wo_pk[0], bo[0], out, x1,
      sT, 2 * sW, sW, DD, HH, WW / 8, sB, sD, sH, 8 * sW);

  // d-pass: 4 sequences along w per block; outer (b, t, h, w/4).
  axial_mha<0, 1><<<BB * TT * HH * (WW / 4), 256, lds, stream>>>(
      x1, qkv_pk[1], bqkv[1], wo_pk[1], bo[1], out, nullptr,
      sD, sW, sW, TT, HH, WW / 4, sB, sT, sH, 4 * sW);

  // h-pass: 4 sequences along w per block; outer (b, t, d, w/4).
  axial_mha<0, 1><<<BB * TT * DD * (WW / 4), 256, lds, stream>>>(
      x1, qkv_pk[2], bqkv[2], wo_pk[2], bo[2], out, nullptr,
      sH, sW, sW, TT, DD, WW / 4, sB, sT, sD, 4 * sW);

  // w-pass: 4 sequences along h per block; outer (b, t, d, h/4).
  axial_mha<0, 1><<<BB * TT * DD * (HH / 4), 256, lds, stream>>>(
      x1, qkv_pk[3], bqkv[3], wo_pk[3], bo[3], out, nullptr,
      sW, sH, sH, TT, DD, HH / 4, sB, sT, sD, 4 * sH);
}
